// Normalizing_Flow_Net_19688130085113
// MI455X (gfx1250) — matrix operand order repaired and checked by reading, not compiled, not run on hardware
//
#include <hip/hip_runtime.h>
#include <hip/hip_bf16.h>
#include <cstdint>
#include <cstddef>

// ---------------------------------------------------------------------------
// Normalizing flow net, MI455X (gfx1250).
//   B=32768, D=16, H=2048, 6 layers.  Dominant cost: 6 x (32768x2048x2048)
//   GEMM = 1.65 TFLOP -> bf16 WMMA (v_wmma_f32_16x16x32_bf16) with f32 acc.
//   W2 is pre-converted AND pre-transposed to bf16 [n][k] once, so both GEMM
//   tiles stage into LDS as contiguous 16B transfers via CDNA5
//   global_load_async_to_lds_b128 (ASYNCcnt), double-buffered so the copies
//   overlap the WMMA pipe.  Batch processed in 8192-row chunks so
//   h1+h2 (bf16) + W2 (bf16) stay resident in the 192MB L2; HBM traffic
//   ~1.5GB total (~65us @ 23.3TB/s) vs ~1ms of WMMA math -> compute-bound.
// ---------------------------------------------------------------------------

#define B_TOTAL   32768
#define D_DIM     16
#define HALF_D    8
#define IN_DIM    11          // HALF + 3
#define H_DIM     2048
#define NUM_LAYERS 6
#define NOISE_SCALE 0.05f

typedef __attribute__((ext_vector_type(16))) __bf16 v16bf;
typedef __attribute__((ext_vector_type(8)))  float  v8f;

// Fixed lane permutation of the D=16 channels, matching
// np.random.default_rng(0).permutation(16) in the reference.
__constant__ int PERM[D_DIM] = {11, 4, 0, 3, 8, 6, 1, 13, 5, 15, 2, 9, 14, 10, 7, 12};

// ------------------------- CDNA5 async-copy helpers -------------------------
// LDS byte offset of a shared-memory pointer: generic (flat) LDS addresses are
// {shared_aperture_hi, offset[31:0]}, and the HW truncates to addr[31:0].
__device__ __forceinline__ unsigned lds_off_of(const void* p) {
    return (unsigned)(size_t)p;
}

// global -> LDS 16-byte async copy, tracked by ASYNCcnt (no VGPR round-trip).
__device__ __forceinline__ void async_copy_b128(unsigned lds_off, const void* gptr) {
    asm volatile("global_load_async_to_lds_b128 %0, %1, off"
                 :: "v"(lds_off), "v"((unsigned long long)(size_t)gptr)
                 : "memory");
}

__device__ __forceinline__ void wait_async0() {
    asm volatile("s_wait_asynccnt 0x0" ::: "memory");
}

// ------------------------- deterministic noise ------------------------------
__device__ __forceinline__ unsigned pcg_hash(unsigned x) {
    x = x * 747796405u + 2891336453u;
    x = ((x >> ((x >> 28u) + 4u)) ^ x) * 277803737u;
    return (x >> 22u) ^ x;
}

__device__ __forceinline__ float gauss_noise(unsigned layer, unsigned row, unsigned k) {
    unsigned s  = pcg_hash(layer * 0x9E3779B9u ^ row * 0x85EBCA6Bu ^ k * 0xC2B2AE35u);
    unsigned ua = pcg_hash(s ^ 0x68BC21EBu);
    unsigned ub = pcg_hash(s ^ 0x02E5BE93u);
    float u1 = (float)(ua + 1u) * 2.3283064e-10f;   // (0,1]
    float u2 = (float)ub        * 2.3283064e-10f;
    return __fsqrt_rn(-2.0f * __logf(u1)) * __cosf(6.2831853f * u2);
}

// ------------------------- setup kernels ------------------------------------
__global__ void zero_ldj_kernel(float* __restrict__ ldj) {
    int i = blockIdx.x * 256 + threadIdx.x;
    if (i < B_TOTAL) ldj[i] = 0.0f;
}

// W2 (f32, [k][n]) -> bf16 transposed [n][k], via LDS 32x32 tile (one-time).
__global__ __launch_bounds__(256) void cvt_w2t_kernel(
    const float* __restrict__ w, __bf16* __restrict__ o)
{
    __shared__ __bf16 tile[32][33];
    const int k0 = blockIdx.x * 32;
    const int n0 = blockIdx.y * 32;
    const int r  = threadIdx.x >> 5;   // 0..7
    const int c  = threadIdx.x & 31;
    #pragma unroll
    for (int i = 0; i < 4; ++i) {
        int kk = r + i * 8;
        tile[kk][c] = (__bf16)w[(size_t)(k0 + kk) * H_DIM + n0 + c];
    }
    __syncthreads();
    #pragma unroll
    for (int i = 0; i < 4; ++i) {
        int nn = r + i * 8;
        o[(size_t)(n0 + nn) * H_DIM + k0 + c] = tile[c][nn];
    }
}

// ----------------- layer front: noise + layernorm + W1 GEMM -----------------
__global__ __launch_bounds__(256) void flow_in_kernel(
    const float* __restrict__ arm, const float* __restrict__ car_x,
    const float* __restrict__ car_y, const float* __restrict__ cvals,
    const float* __restrict__ W1, const float* __restrict__ b1,
    __bf16* __restrict__ h1, int layer, int rowBase)
{
    __shared__ float ci[IN_DIM];
    __shared__ float stat[2];
    const int t   = threadIdx.x;
    const int row = rowBase + blockIdx.x;

    if (t < IN_DIM) {
        float v;
        if      (t < HALF_D) v = arm[(size_t)row * D_DIM + t];
        else if (t == 8)     v = car_x[row];
        else if (t == 9)     v = car_y[row];
        else                 v = cvals[row];
        ci[t] = v + NOISE_SCALE * gauss_noise((unsigned)layer, (unsigned)row, (unsigned)t);
    }
    __syncthreads();
    if (t == 0) {
        float mu = 0.0f;
        #pragma unroll
        for (int k = 0; k < IN_DIM; ++k) mu += ci[k];
        mu *= (1.0f / IN_DIM);
        float var = 0.0f;
        #pragma unroll
        for (int k = 0; k < IN_DIM; ++k) { float d = ci[k] - mu; var += d * d; }
        var *= (1.0f / (IN_DIM - 1));               // ddof=1
        stat[0] = mu;
        stat[1] = __frsqrt_rn(var);
    }
    __syncthreads();

    float x[IN_DIM];
    const float mu = stat[0], inv = stat[1];
    #pragma unroll
    for (int k = 0; k < IN_DIM; ++k) x[k] = (ci[k] - mu) * inv;

    __bf16* hrow = h1 + (size_t)blockIdx.x * H_DIM;
    #pragma unroll
    for (int jj = 0; jj < H_DIM / 256; ++jj) {
        int j = t + jj * 256;
        float acc = b1[j];
        #pragma unroll
        for (int k = 0; k < IN_DIM; ++k)
            acc = fmaf(x[k], W1[k * H_DIM + j], acc);
        hrow[j] = (__bf16)fmaxf(acc, 0.0f);
    }
}

// ----------------- layer core: h2 = relu(h1 @ W2 + b2) ----------------------
// 128x128x(K=32) tiles, 8 waves/block, each wave = 4x2 WMMA tiles (64x32).
// Both tiles live in LDS as [row][k] (W2 pre-transposed), double-buffered and
// filled with global_load_async_to_lds_b128; fragments are contiguous
// ds_load_b128 pairs matching the CDNA5 16-bit WMMA VGPR layout.
__global__ __launch_bounds__(256, 2) void flow_gemm_kernel(
    const __bf16* __restrict__ h1, const __bf16* __restrict__ w2t,
    const float* __restrict__ b2, __bf16* __restrict__ h2)
{
    __shared__ __align__(16) __bf16 lA[2][128 * 32];   // [m][k]
    __shared__ __align__(16) __bf16 lB[2][128 * 32];   // [n][k]

    const int t    = threadIdx.x;
    const int m0   = blockIdx.x * 128;
    const int n0   = blockIdx.y * 128;
    const int wave = t >> 5;
    const int lane = t & 31;
    const int wm   = (wave & 1) * 64;   // 2 waves along M
    const int wn   = (wave >> 1) * 32;  // 4 waves along N
    const int l15  = lane & 15;
    const int lh   = lane >> 4;

    // ---- per-thread staging slots: 2 A + 2 B transfers of 16B per tile
    const int ar0 = t >> 2,          ac0 = (t & 3) * 8;
    const int ar1 = (t + 256) >> 2,  ac1 = ((t + 256) & 3) * 8;
    const __bf16* gA0 = h1  + (size_t)(m0 + ar0) * H_DIM + ac0;
    const __bf16* gA1 = h1  + (size_t)(m0 + ar1) * H_DIM + ac1;
    const __bf16* gB0 = w2t + (size_t)(n0 + ar0) * H_DIM + ac0;
    const __bf16* gB1 = w2t + (size_t)(n0 + ar1) * H_DIM + ac1;
    unsigned laA0[2], laA1[2], laB0[2], laB1[2];
    #pragma unroll
    for (int b = 0; b < 2; ++b) {
        laA0[b] = lds_off_of(&lA[b][ar0 * 32 + ac0]);
        laA1[b] = lds_off_of(&lA[b][ar1 * 32 + ac1]);
        laB0[b] = lds_off_of(&lB[b][ar0 * 32 + ac0]);
        laB1[b] = lds_off_of(&lB[b][ar1 * 32 + ac1]);
    }
    auto issue_tile = [&](int buf, int k0) {
        async_copy_b128(laA0[buf], gA0 + k0);
        async_copy_b128(laA1[buf], gA1 + k0);
        async_copy_b128(laB0[buf], gB0 + k0);
        async_copy_b128(laB1[buf], gB1 + k0);
    };

    const v8f vzero = {0.f, 0.f, 0.f, 0.f, 0.f, 0.f, 0.f, 0.f};
    v8f acc[4][2];
    #pragma unroll
    for (int mi = 0; mi < 4; ++mi)
        #pragma unroll
        for (int ni = 0; ni < 2; ++ni) acc[mi][ni] = vzero;

    union FragU { uint4 u[2]; v16bf v; };

    const int NT = H_DIM / 32;
    issue_tile(0, 0);
    for (int kt = 0; kt < NT; ++kt) {
        wait_async0();          // this thread's copies for tile kt done
        __syncthreads();        // whole tile kt visible; tile kt-1 compute done
        if (kt + 1 < NT) issue_tile((kt + 1) & 1, (kt + 1) * 32);

        const __bf16* A  = lA[kt & 1];
        const __bf16* Bm = lB[kt & 1];

        // A fragments: lane m=l15, K = lh*8 + {0..7, 16..23}
        FragU af[4], bfr[2];
        #pragma unroll
        for (int mi = 0; mi < 4; ++mi) {
            int mr = wm + mi * 16 + l15;
            af[mi].u[0] = *(const uint4*)&A[mr * 32 + lh * 8];
            af[mi].u[1] = *(const uint4*)&A[mr * 32 + lh * 8 + 16];
        }
        // B fragments: lane n=l15, K = lh*8 + {0..7, 16..23} in [n][k]
        #pragma unroll
        for (int ni = 0; ni < 2; ++ni) {
            int nr = wn + ni * 16 + l15;
            bfr[ni].u[0] = *(const uint4*)&Bm[nr * 32 + lh * 8];
            bfr[ni].u[1] = *(const uint4*)&Bm[nr * 32 + lh * 8 + 16];
        }
        #pragma unroll
        for (int mi = 0; mi < 4; ++mi)
            #pragma unroll
            for (int ni = 0; ni < 2; ++ni)
                acc[mi][ni] = __builtin_amdgcn_wmma_f32_16x16x32_bf16(
                    false, af[mi].v, false, bfr[ni].v,
                    (short)0, acc[mi][ni], false, false);
    }

    // ---- epilogue: bias + relu + bf16 store
    #pragma unroll
    for (int mi = 0; mi < 4; ++mi) {
        #pragma unroll
        for (int ni = 0; ni < 2; ++ni) {
            int n = n0 + wn + ni * 16 + l15;
            float bias = b2[n];
            #pragma unroll
            for (int r = 0; r < 8; ++r) {
                int m = m0 + wm + mi * 16 + lh * 8 + r;
                float v = acc[mi][ni][r] + bias;
                h2[(size_t)m * H_DIM + n] = (__bf16)fmaxf(v, 0.0f);
            }
        }
    }
}

// ----------------- layer tail: W3 GEMM + coupling epilogue ------------------
__global__ __launch_bounds__(256) void flow_out_kernel(
    const __bf16* __restrict__ h2, const float* __restrict__ W3,
    const float* __restrict__ b3, float* __restrict__ arm,
    float* __restrict__ ldj, int rowBase)
{
    __shared__ float outv[16][16];
    __shared__ float nrow[16][17];
    __shared__ float lv[16][8];

    const int t    = threadIdx.x;
    const int rl   = t >> 4;
    const int j    = t & 15;
    const int lrow = blockIdx.x * 16 + rl;
    const int row  = rowBase + lrow;

    const __bf16* hr = h2 + (size_t)lrow * H_DIM;
    float acc = b3[j];
    for (int k = 0; k < H_DIM; k += 8) {
        union { uint4 u; __bf16 h[8]; } cv;
        cv.u = *(const uint4*)(hr + k);
        #pragma unroll
        for (int e = 0; e < 8; ++e)
            acc = fmaf((float)cv.h[e], W3[(k + e) * D_DIM + j], acc);
    }
    outv[rl][j] = acc;
    __syncthreads();

    if (j < HALF_D) {
        float s  = fminf(10.0f, fmaxf(-10.0f, outv[rl][j]));
        float tt = fminf(10.0f, fmaxf(-10.0f, outv[rl][j + HALF_D]));
        float alt = arm[(size_t)row * D_DIM + HALF_D + j];
        nrow[rl][HALF_D + j] = alt * __expf(s) + tt;
        nrow[rl][j]          = arm[(size_t)row * D_DIM + j];   // const half
        lv[rl][j]            = __logf(fabsf(s));
    }
    __syncthreads();

    if (j == 0) {
        float ssum = 0.0f;
        #pragma unroll
        for (int k = 0; k < HALF_D; ++k) ssum += lv[rl][k];
        ldj[row] += ssum * (1.0f / HALF_D);
    }
    arm[(size_t)row * D_DIM + j] = nrow[rl][PERM[j]];
}

// ---------------------------------------------------------------------------
extern "C" void kernel_launch(void* const* d_in, const int* in_sizes, int n_in,
                              void* d_out, int out_size, void* d_ws, size_t ws_size,
                              hipStream_t stream)
{
    (void)in_sizes; (void)n_in; (void)out_size;
    const float* arm_in = (const float*)d_in[0];
    const float* car_x  = (const float*)d_in[1];
    const float* car_y  = (const float*)d_in[2];
    const float* cvals  = (const float*)d_in[3];
    const float* W1     = (const float*)d_in[4];
    const float* b1     = (const float*)d_in[5];
    const float* W2     = (const float*)d_in[6];
    const float* b2     = (const float*)d_in[7];
    const float* W3     = (const float*)d_in[8];
    const float* b3     = (const float*)d_in[9];

    float* arm = (float*)d_out;
    float* ldj = arm + (size_t)B_TOTAL * D_DIM;

    // arm evolves in place in d_out across layers; ldj accumulates from 0.
    hipMemcpyAsync(arm, arm_in, (size_t)B_TOTAL * D_DIM * sizeof(float),
                   hipMemcpyDeviceToDevice, stream);
    zero_ldj_kernel<<<B_TOTAL / 256, 256, 0, stream>>>(ldj);

    // Workspace: [ W2^T bf16 8MB | h1 chunk bf16 | h2 chunk bf16 ]
    const size_t W2_BYTES = (size_t)H_DIM * H_DIM * 2;
    __bf16* w2t = (__bf16*)d_ws;
    {
        dim3 g(H_DIM / 32, H_DIM / 32);
        cvt_w2t_kernel<<<g, 256, 0, stream>>>(W2, w2t);
    }

    size_t avail = ws_size > W2_BYTES ? ws_size - W2_BYTES : 0;
    long long chunk = (long long)(avail / ((size_t)2 * H_DIM * 2));
    chunk = (chunk / 128) * 128;
    if (chunk > 8192)    chunk = 8192;   // keep h1+h2+W2t resident in 192MB L2
    if (chunk < 128)     chunk = 128;    // minimum tile; assumes ws >= ~9MB
    if (chunk > B_TOTAL) chunk = B_TOTAL;

    __bf16* h1 = (__bf16*)((char*)d_ws + W2_BYTES);
    __bf16* h2 = h1 + (size_t)chunk * H_DIM;

    for (int base = 0; base < B_TOTAL; base += (int)chunk) {
        int rem = B_TOTAL - base;
        int cur = rem < (int)chunk ? rem : (int)chunk;
        for (int layer = 0; layer < NUM_LAYERS; ++layer) {
            flow_in_kernel<<<cur, 256, 0, stream>>>(
                arm, car_x, car_y, cvals, W1, b1, h1, layer, base);
            dim3 gB(cur / 128, H_DIM / 128);
            flow_gemm_kernel<<<gB, 256, 0, stream>>>(h1, w2t, b2, h2);
            flow_out_kernel<<<cur / 16, 256, 0, stream>>>(
                h2, W3, b3, arm, ldj, base);
        }
    }
}
